// DentalPointTransformer_11793980195206
// MI455X (gfx1250) — matrix operand order repaired and checked by reading, not compiled, not run on hardware
//
#include <hip/hip_runtime.h>
#include <math.h>

// ---------------------------------------------------------------------------
// DentalPointTransformer forward for MI455X (gfx1250, wave32, WMMA).
// GEMMs run on v_wmma_f32_16x16x32_bf16 with pure-bf16 operand streams:
// weights pre-converted once, activations shadowed in bf16 by the producer
// kernels, so the GEMM inner loop is loads + WMMA only (no converts).
// ---------------------------------------------------------------------------

typedef __attribute__((ext_vector_type(16))) __bf16 v16bf;
typedef __attribute__((ext_vector_type(8)))  __bf16 v8bf;
typedef __attribute__((ext_vector_type(8)))  float  v8f;
typedef __attribute__((ext_vector_type(4)))  float  v4f;

#define NTOT  32768
#define NPTS_ 4096

__device__ __forceinline__ float wave_sum_f(float v) {
#pragma unroll
  for (int o = 16; o > 0; o >>= 1) v += __shfl_xor(v, o, 32);
  return v;
}
__device__ __forceinline__ float wave_max_f(float v) {
#pragma unroll
  for (int o = 16; o > 0; o >>= 1) v = fmaxf(v, __shfl_xor(v, o, 32));
  return v;
}

// ---------------------------------------------------------------------------
// KNN: one wave per point. Per-lane sorted top-16 in registers, then a
// shuffle-argmin merge of the 32 sorted lists through LDS.
// ---------------------------------------------------------------------------
__global__ __launch_bounds__(128) void knn_kernel(const float* __restrict__ pos,
                                                  int* __restrict__ nbr) {
  const int w    = threadIdx.x >> 5;
  const int lane = threadIdx.x & 31;
  const int i    = blockIdx.x * 4 + w;
  const int cbase = (i / NPTS_) * NPTS_;

  const float* pi = pos + (size_t)i * 6;
  const float p0 = pi[0], p1 = pi[1], p2 = pi[2], p3 = pi[3], p4 = pi[4], p5 = pi[5];

  float bd[16]; int bj[16];
#pragma unroll
  for (int s = 0; s < 16; ++s) { bd[s] = 1e30f; bj[s] = 0; }

  for (int jj = lane; jj < NPTS_; jj += 32) {
    const int j = cbase + jj;
    if (j == i) continue;                       // exclude self (diag +1e10)
    const float* q = pos + (size_t)j * 6;
    float d0 = q[0] - p0, d1 = q[1] - p1, d2 = q[2] - p2;
    float d3 = q[3] - p3, d4 = q[4] - p4, d5 = q[5] - p5;
    float d = d0*d0 + d1*d1 + d2*d2 + d3*d3 + d4*d4 + d5*d5;
    if (d < bd[15]) {                           // branch-free sorted insert
#pragma unroll
      for (int s = 15; s >= 0; --s) {
        bool shift = (s > 0) && (bd[s - 1] > d);
        float nd = shift ? bd[s - 1] : d;
        int   nj = shift ? bj[s - 1] : j;
        if (bd[s] > d) { bd[s] = nd; bj[s] = nj; }
      }
    }
  }

  __shared__ float sd[4][32][16];
  __shared__ int   sj[4][32][16];
#pragma unroll
  for (int s = 0; s < 16; ++s) { sd[w][lane][s] = bd[s]; sj[w][lane][s] = bj[s]; }
  __syncthreads();

  int head = 0;
  for (int r = 0; r < 16; ++r) {
    float v  = (head < 16) ? sd[w][lane][head] : 1e30f;
    int   jv = (head < 16) ? sj[w][lane][head] : 0;
    int   sl = lane;
#pragma unroll
    for (int o = 16; o > 0; o >>= 1) {
      float ov  = __shfl_xor(v, o, 32);
      int   ojv = __shfl_xor(jv, o, 32);
      int   osl = __shfl_xor(sl, o, 32);
      if (ov < v || (ov == v && osl < sl)) { v = ov; jv = ojv; sl = osl; }
    }
    if (lane == sl) head++;
    if (lane == 0) nbr[(size_t)i * 17 + r] = jv;
  }
  if (lane == 0) nbr[(size_t)i * 17 + 16] = i;
}

// ---------------------------------------------------------------------------
// f32 -> bf16 copy (weight pre-conversion)
// ---------------------------------------------------------------------------
__global__ __launch_bounds__(256) void cvt_bf16_kernel(const float* __restrict__ src,
                                                       __bf16* __restrict__ dst,
                                                       size_t total) {
  size_t i = (size_t)blockIdx.x * 256 + threadIdx.x;
  if (i < total) dst[i] = (__bf16)src[i];
}

// ---------------------------------------------------------------------------
// lin0: y[N,64] = pos[N,6] @ W[64,6]^T + b  (f32 + bf16 shadow)
// ---------------------------------------------------------------------------
__global__ __launch_bounds__(256) void lin0_kernel(const float* __restrict__ pos,
                                                   const float* __restrict__ w,
                                                   const float* __restrict__ b,
                                                   float* __restrict__ y,
                                                   __bf16* __restrict__ ybf) {
  int idx = blockIdx.x * 256 + threadIdx.x;   // over N*64
  int i = idx >> 6, c = idx & 63;
  const float* p = pos + (size_t)i * 6;
  const float* wr = w + (size_t)c * 6;
  float acc = b[c];
#pragma unroll
  for (int d = 0; d < 6; ++d) acc += p[d] * wr[d];
  y[idx] = acc;
  ybf[idx] = (__bf16)acc;
}

// ---------------------------------------------------------------------------
// WMMA GEMM: Y[n,cout] = act(X[n,cin] @ W[cout,cin]^T + bias)
// Pure bf16 operands, f32 accumulation, software-pipelined K loop.
// Block = 8 waves; each wave owns a 32(M) x 64(N) tile (8 accumulators,
// B fragments reused across both row halves). grid = (cout/64, n/256).
// Requires cin % 32 == 0, cout % 64 == 0, n % 256 == 0.
// ---------------------------------------------------------------------------
__device__ __forceinline__ v16bf loadA16(const __bf16* p) {
  return *(const v16bf*)p;            // K base .. base+15
}

__global__ __launch_bounds__(256) void gemm_wmma_kernel(
    const __bf16* __restrict__ X, const __bf16* __restrict__ W,
    const float* __restrict__ bias, float* __restrict__ Yf,
    __bf16* __restrict__ Ybf, int n, int cin, int cout, int do_relu) {
  const int lane = threadIdx.x & 31;
  const int wave = threadIdx.x >> 5;
  const int row0 = blockIdx.y * 256 + wave * 32;
  const int col0 = blockIdx.x * 64;
  const int lr = lane & 15;   // row within A tile / col within B subtile
  const int lh = lane >> 4;   // lane-half selector

  const __bf16* xr0 = X + (size_t)(row0 + lr) * cin + lh * 16;
  const __bf16* xr1 = xr0 + (size_t)16 * cin;
  const __bf16* wp0 = W + (size_t)(col0 + 0  + lr) * cin + lh * 16;
  const __bf16* wp1 = W + (size_t)(col0 + 16 + lr) * cin + lh * 16;
  const __bf16* wp2 = W + (size_t)(col0 + 32 + lr) * cin + lh * 16;
  const __bf16* wp3 = W + (size_t)(col0 + 48 + lr) * cin + lh * 16;

  v8f acc[2][4] = {};

  // prologue loads (K block 0)
  v16bf ca0 = loadA16(xr0);
  v16bf ca1 = loadA16(xr1);
  v16bf cb0 = *(const v16bf*)wp0;
  v16bf cb1 = *(const v16bf*)wp1;
  v16bf cb2 = *(const v16bf*)wp2;
  v16bf cb3 = *(const v16bf*)wp3;

  for (int k = 0; k < cin; k += 32) {
    v16bf na0 = ca0, na1 = ca1, nb0 = cb0, nb1 = cb1, nb2 = cb2, nb3 = cb3;
    if (k + 32 < cin) {         // issue next-tile loads before consuming regs
      na0 = loadA16(xr0 + k + 32);
      na1 = loadA16(xr1 + k + 32);
      nb0 = *(const v16bf*)(wp0 + k + 32);
      nb1 = *(const v16bf*)(wp1 + k + 32);
      nb2 = *(const v16bf*)(wp2 + k + 32);
      nb3 = *(const v16bf*)(wp3 + k + 32);
    }
    acc[0][0] = __builtin_amdgcn_wmma_f32_16x16x32_bf16(false, ca0, false, cb0, (short)0, acc[0][0], false, false);
    acc[1][0] = __builtin_amdgcn_wmma_f32_16x16x32_bf16(false, ca1, false, cb0, (short)0, acc[1][0], false, false);
    acc[0][1] = __builtin_amdgcn_wmma_f32_16x16x32_bf16(false, ca0, false, cb1, (short)0, acc[0][1], false, false);
    acc[1][1] = __builtin_amdgcn_wmma_f32_16x16x32_bf16(false, ca1, false, cb1, (short)0, acc[1][1], false, false);
    acc[0][2] = __builtin_amdgcn_wmma_f32_16x16x32_bf16(false, ca0, false, cb2, (short)0, acc[0][2], false, false);
    acc[1][2] = __builtin_amdgcn_wmma_f32_16x16x32_bf16(false, ca1, false, cb2, (short)0, acc[1][2], false, false);
    acc[0][3] = __builtin_amdgcn_wmma_f32_16x16x32_bf16(false, ca0, false, cb3, (short)0, acc[0][3], false, false);
    acc[1][3] = __builtin_amdgcn_wmma_f32_16x16x32_bf16(false, ca1, false, cb3, (short)0, acc[1][3], false, false);
    ca0 = na0; ca1 = na1; cb0 = nb0; cb1 = nb1; cb2 = nb2; cb3 = nb3;
  }

  // Epilogue: C/D layout -> M = r + lh*8 (+16 for second half), N = lr.
#pragma unroll
  for (int half = 0; half < 2; ++half) {
#pragma unroll
    for (int t = 0; t < 4; ++t) {
      const int col = col0 + t * 16 + lr;
      const float bv = bias[col];
#pragma unroll
      for (int r = 0; r < 8; ++r) {
        const int row = row0 + half * 16 + r + lh * 8;
        float v = acc[half][t][r] + bv;
        if (do_relu) v = fmaxf(v, 0.0f);
        if (Yf)  Yf[(size_t)row * cout + col] = v;
        if (Ybf) Ybf[(size_t)row * cout + col] = (__bf16)v;
      }
    }
  }
}

// ---------------------------------------------------------------------------
// GATv2 attention: one wave per node. Lanes 0..16 own edges (logits+softmax),
// then lanes stripe channels for the alpha-weighted gather.
// ---------------------------------------------------------------------------
__global__ __launch_bounds__(128) void gatv2_kernel(
    const float* __restrict__ xl, const float* __restrict__ xr,
    const int* __restrict__ nbr, const float* __restrict__ att,
    const float* __restrict__ bias, float* __restrict__ out, int cout) {
  const int w = threadIdx.x >> 5;
  const int lane = threadIdx.x & 31;
  const int i = blockIdx.x * 4 + w;
  const int ch = cout >> 2;   // HEADS = 4

  __shared__ int   s_nbr[4][17];
  __shared__ float s_alpha[4][17 * 4];

  if (lane < 17) s_nbr[w][lane] = nbr[(size_t)i * 17 + lane];
  __syncthreads();

  float lg[4] = {-1e30f, -1e30f, -1e30f, -1e30f};
  if (lane < 17) {
    const int j = s_nbr[w][lane];
    const float* pj = xl + (size_t)j * cout;
    const float* pi = xr + (size_t)i * cout;
#pragma unroll
    for (int h = 0; h < 4; ++h) {
      float acc = 0.0f;
      const float* ah  = att + h * ch;
      const float* pjh = pj + h * ch;
      const float* pih = pi + h * ch;
      for (int c = 0; c < ch; ++c) {
        float e = pjh[c] + pih[c];
        e = (e > 0.0f) ? e : 0.2f * e;          // leaky_relu(0.2)
        acc += ah[c] * e;
      }
      lg[h] = acc;
    }
  }
#pragma unroll
  for (int h = 0; h < 4; ++h) {
    float m = wave_max_f(lg[h]);
    float ev = (lane < 17) ? expf(lg[h] - m) : 0.0f;
    float ssum = wave_sum_f(ev);
    if (lane < 17) s_alpha[w][lane * 4 + h] = ev / ssum;
  }
  __syncthreads();

  for (int c = lane; c < cout; c += 32) {
    const int h = c / ch;
    float acc = 0.0f;
#pragma unroll
    for (int k = 0; k < 17; ++k) {
      acc += s_alpha[w][k * 4 + h] * xl[(size_t)s_nbr[w][k] * cout + c];
    }
    out[(size_t)i * cout + c] = acc + bias[c];
  }
}

// ---------------------------------------------------------------------------
// BatchNorm (training stats, biased variance): stats + apply (+bf16 shadow).
// ---------------------------------------------------------------------------
__global__ __launch_bounds__(256) void bn_stats_kernel(const float* __restrict__ x,
                                                       int n, int c,
                                                       float* __restrict__ mean,
                                                       float* __restrict__ rstd) {
  const int ch = blockIdx.x;
  const int t = threadIdx.x;
  float s = 0.0f, s2 = 0.0f;
  for (int r = t; r < n; r += 256) {
    float v = x[(size_t)r * c + ch];
    s += v; s2 += v * v;
  }
  __shared__ float sh0[256], sh1[256];
  sh0[t] = s; sh1[t] = s2;
  __syncthreads();
  for (int o = 128; o > 0; o >>= 1) {
    if (t < o) { sh0[t] += sh0[t + o]; sh1[t] += sh1[t + o]; }
    __syncthreads();
  }
  if (t == 0) {
    float m = sh0[0] / (float)n;
    float var = sh1[0] / (float)n - m * m;
    mean[ch] = m;
    rstd[ch] = rsqrtf(var + 1e-5f);
  }
}

__global__ __launch_bounds__(256) void bn_apply_kernel(
    const float* __restrict__ x, const float* __restrict__ mean,
    const float* __restrict__ rstd, const float* __restrict__ g,
    const float* __restrict__ b, float* __restrict__ y,
    __bf16* __restrict__ ybf, int cmask, size_t total) {
  size_t idx = (size_t)blockIdx.x * 256 + threadIdx.x;
  if (idx >= total) return;
  int ch = (int)(idx & (size_t)cmask);
  float v = (x[idx] - mean[ch]) * rstd[ch] * g[ch] + b[ch];
  y[idx] = v;
  if (ybf) ybf[idx] = (__bf16)v;
}

__global__ __launch_bounds__(256) void add_kernel(const float* __restrict__ a,
                                                  const float* __restrict__ b,
                                                  float* __restrict__ y, size_t total) {
  size_t idx = (size_t)blockIdx.x * 256 + threadIdx.x;
  if (idx < total) y[idx] = a[idx] + b[idx];
}

__global__ __launch_bounds__(256) void concat_kernel(const float* __restrict__ x1,
                                                     const float* __restrict__ x3,
                                                     const float* __restrict__ x4,
                                                     __bf16* __restrict__ y) {
  size_t idx = (size_t)blockIdx.x * 256 + threadIdx.x;  // over N*448
  size_t i = idx / 448;
  int j = (int)(idx % 448);
  float v;
  if (j < 64)       v = x1[i * 64 + j];
  else if (j < 192) v = x3[i * 128 + (j - 64)];
  else              v = x4[i * 256 + (j - 192)];
  y[idx] = (__bf16)v;
}

// ---------------------------------------------------------------------------
// ArcFace: normalize arc_w rows (3x128) once, then per-node logits.
// ---------------------------------------------------------------------------
__global__ __launch_bounds__(128) void normw_kernel(const float* __restrict__ w,
                                                    float* __restrict__ wn) {
  const int wv = threadIdx.x >> 5;
  const int lane = threadIdx.x & 31;
  if (wv >= 3) return;
  v4f v = ((const v4f*)(w + wv * 128))[lane];
  float s = v[0]*v[0] + v[1]*v[1] + v[2]*v[2] + v[3]*v[3];
  s = wave_sum_f(s);
  float r = rsqrtf(s);
  v4f o; o[0] = v[0]*r; o[1] = v[1]*r; o[2] = v[2]*r; o[3] = v[3]*r;
  ((v4f*)(wn + wv * 128))[lane] = o;
}

__global__ __launch_bounds__(128) void arcface_kernel(const float* __restrict__ emb,
                                                      const float* __restrict__ wn,
                                                      const int* __restrict__ label,
                                                      float* __restrict__ out) {
  const int w = threadIdx.x >> 5;
  const int lane = threadIdx.x & 31;
  const int i = blockIdx.x * 4 + w;
  v4f e = ((const v4f*)(emb + (size_t)i * 128))[lane];
  float s = wave_sum_f(e[0]*e[0] + e[1]*e[1] + e[2]*e[2] + e[3]*e[3]);
  float inv = rsqrtf(s);                        // 1/||emb||
  float cosv[3];
#pragma unroll
  for (int c = 0; c < 3; ++c) {
    v4f wv = ((const v4f*)(wn + c * 128))[lane];
    float p = e[0]*wv[0] + e[1]*wv[1] + e[2]*wv[2] + e[3]*wv[3];
    cosv[c] = wave_sum_f(p) * inv;
  }
  if (lane < 3) {
    float cc = cosv[lane];
    float thc = fminf(fmaxf(cc, -1.0f + 1e-7f), 1.0f - 1e-7f);
    float tl = cosf(acosf(thc) + 0.5f);         // cos(theta + m)
    int lb = label[i];
    out[(size_t)i * 3 + lane] = ((lane == lb) ? tl : cc) * 30.0f;
  }
}

// ---------------------------------------------------------------------------
// Orchestration
// ---------------------------------------------------------------------------
extern "C" void kernel_launch(void* const* d_in, const int* in_sizes, int n_in,
                              void* d_out, int out_size, void* d_ws, size_t ws_size,
                              hipStream_t stream) {
  (void)in_sizes; (void)out_size;
  if (n_in < 66) return;

  const int Np = NTOT;
  const float* pos    = (const float*)d_in[0];
  /* d_in[1] = batch (unused: block-contiguous, baked into knn) */
  const int*   label  = (const int*)d_in[2];   // jax default x64-off -> int32
  const float* lin0_w = (const float*)d_in[3];
  const float* lin0_b = (const float*)d_in[4];
  // per-layer params, insertion order:
  // 0:wl 1:bl 2:wr 3:br 4:att 5:bias 6:ffn_w1 7:ffn_b1 8:ffn_w2 9:ffn_b2
  // 10:bn1_g 11:bn1_b 12:bn2_g 13:bn2_b
  const float* LP[4][14];
  for (int l = 0; l < 4; ++l)
    for (int q = 0; q < 14; ++q)
      LP[l][q] = (const float*)d_in[5 + l * 14 + q];
  const float* head_w1 = (const float*)d_in[61];
  const float* head_b1 = (const float*)d_in[62];
  const float* head_w2 = (const float*)d_in[63];
  const float* head_b2 = (const float*)d_in[64];
  const float* arc_w   = (const float*)d_in[65];

  // ---------------- workspace layout (~250 MB) ----------------
  char* ws = (char*)d_ws;
  size_t o = 0;
  auto take = [&o](size_t bytes) {
    size_t r = o; o += (bytes + 255) & ~(size_t)255; return r;
  };
  const size_t off_nbr    = take((size_t)Np * 17 * 4);
  const size_t off_x1     = take((size_t)Np * 64 * 4);
  const size_t off_x3     = take((size_t)Np * 128 * 4);
  const size_t off_stats  = take(1024 * 4);
  const size_t off_wn     = take(384 * 4);
  const size_t off_wbf    = take(650000 * 2);           // bf16 weight arena
  const size_t off_xaltbf = take((size_t)Np * 256 * 2); // x0bf / x2bf
  const size_t off_x1bf   = take((size_t)Np * 64 * 2);
  const size_t off_x3bf   = take((size_t)Np * 128 * 2);
  const size_t off_xbnbf  = take((size_t)Np * 256 * 2); // BN1 output shadow
  const size_t off_hidbf  = take((size_t)Np * 256 * 2); // FFN hidden (bf16 only)
  const size_t off_xalt   = take((size_t)Np * 256 * 4); // x0 / x2 / emb (f32)
  const size_t off_x4     = take((size_t)Np * 256 * 4);
  const size_t off_C      = take((size_t)Np * 256 * 4); // scratch C
  const size_t off_A      = take((size_t)Np * 256 * 4); // scratch A / comb_bf
  const size_t off_B      = take((size_t)Np * 256 * 4); // scratch B / h512_bf
  if (o > ws_size) return;  // workspace too small: nothing safe to do

  int*    nbr   = (int*)(ws + off_nbr);
  float*  x1    = (float*)(ws + off_x1);
  float*  x3    = (float*)(ws + off_x3);
  float*  mean  = (float*)(ws + off_stats);
  float*  rstd  = mean + 512;
  float*  wn    = (float*)(ws + off_wn);
  __bf16* wbf   = (__bf16*)(ws + off_wbf);
  __bf16* xaltbf= (__bf16*)(ws + off_xaltbf);
  __bf16* x1bf  = (__bf16*)(ws + off_x1bf);
  __bf16* x3bf  = (__bf16*)(ws + off_x3bf);
  __bf16* xbnbf = (__bf16*)(ws + off_xbnbf);
  __bf16* hidbf = (__bf16*)(ws + off_hidbf);
  float*  xalt  = (float*)(ws + off_xalt);
  float*  x4    = (float*)(ws + off_x4);
  float*  bufA  = (float*)(ws + off_A);
  float*  bufB  = (float*)(ws + off_B);
  float*  bufC  = (float*)(ws + off_C);
  float*  x0    = xalt;                 // alive only during block 0
  float*  x2    = xalt;                 // alive only during blocks 1-2
  float*  emb   = xalt;                 // alive after head gemm2
  __bf16* combbf= (__bf16*)(ws + off_A); // N*448 bf16 (28MB) in dead scratch A
  __bf16* h512bf= (__bf16*)(ws + off_B); // N*512 bf16 (32MB) in dead scratch B

  // ---- pre-convert weights to bf16 (wl, wr, fw1, fw2 x4; head_w1/2) ----
  size_t wo = 0;
  auto wtake = [&wo](size_t elems) {
    size_t r = wo; wo += (elems + 127) & ~(size_t)127; return r;
  };
  const int CIN[4]  = {64, 64, 128, 128};
  const int COUT[4] = {64, 128, 128, 256};
  __bf16 *wl_bf[4], *wr_bf[4], *f1_bf[4], *f2_bf[4];
  for (int l = 0; l < 4; ++l) {
    size_t e1 = (size_t)COUT[l] * CIN[l];
    size_t e2 = (size_t)COUT[l] * COUT[l];
    wl_bf[l] = wbf + wtake(e1);
    wr_bf[l] = wbf + wtake(e1);
    f1_bf[l] = wbf + wtake(e2);
    f2_bf[l] = wbf + wtake(e2);
  }
  __bf16* hw1_bf = wbf + wtake((size_t)512 * 448);
  __bf16* hw2_bf = wbf + wtake((size_t)128 * 512);
  auto cvt = [&](const float* src, __bf16* dst, size_t elems) {
    cvt_bf16_kernel<<<(unsigned)((elems + 255) / 256), 256, 0, stream>>>(src, dst, elems);
  };
  for (int l = 0; l < 4; ++l) {
    size_t e1 = (size_t)COUT[l] * CIN[l];
    size_t e2 = (size_t)COUT[l] * COUT[l];
    cvt(LP[l][0], wl_bf[l], e1);
    cvt(LP[l][2], wr_bf[l], e1);
    cvt(LP[l][6], f1_bf[l], e2);
    cvt(LP[l][8], f2_bf[l], e2);
  }
  cvt(head_w1, hw1_bf, (size_t)512 * 448);
  cvt(head_w2, hw2_bf, (size_t)128 * 512);

  auto gemm = [&](const __bf16* X, const __bf16* W, const float* bias,
                  float* Yf, __bf16* Ybf, int cin, int cout, int relu) {
    dim3 gg(cout / 64, Np / 256);
    gemm_wmma_kernel<<<gg, 256, 0, stream>>>(X, W, bias, Yf, Ybf,
                                             Np, cin, cout, relu);
  };

  auto run_block = [&](const float* Xf, const __bf16* Xbf, int cin, int cout,
                       const float* const* P, int layer,
                       float* xoutf, __bf16* xoutbf) {
    size_t tot = (size_t)Np * cout;
    unsigned eb = (unsigned)((tot + 255) / 256);
    // xl = X@Wl^T + bl ; xr = X@Wr^T + br (f32, consumed by attention)
    gemm(Xbf, wl_bf[layer], P[1], bufA, nullptr, cin, cout, 0);
    gemm(Xbf, wr_bf[layer], P[3], bufB, nullptr, cin, cout, 0);
    // GATv2 attention (+ attn bias)
    gatv2_kernel<<<Np / 4, 128, 0, stream>>>(bufA, bufB, nbr, P[4], P[5], bufC, cout);
    if (cin == cout)  // residual xa + x
      add_kernel<<<eb, 256, 0, stream>>>(bufC, Xf, bufC, tot);
    // BN1 -> bufB (f32 xbn) + xbnbf (bf16 shadow for FFN gemm)
    bn_stats_kernel<<<cout, 256, 0, stream>>>(bufC, Np, cout, mean, rstd);
    bn_apply_kernel<<<eb, 256, 0, stream>>>(bufC, mean, rstd, P[10], P[11],
                                            bufB, xbnbf, cout - 1, tot);
    // FFN: relu(xbn@W1^T+b1) (bf16 only) @ W2^T + b2 (f32), + xbn, BN2 -> xout
    gemm(xbnbf, f1_bf[layer], P[7], nullptr, hidbf, cout, cout, 1);
    gemm(hidbf, f2_bf[layer], P[9], bufC, nullptr, cout, cout, 0);
    add_kernel<<<eb, 256, 0, stream>>>(bufC, bufB, bufC, tot);
    bn_stats_kernel<<<cout, 256, 0, stream>>>(bufC, Np, cout, mean, rstd);
    bn_apply_kernel<<<eb, 256, 0, stream>>>(bufC, mean, rstd, P[12], P[13],
                                            xoutf, xoutbf, cout - 1, tot);
  };

  knn_kernel<<<Np / 4, 128, 0, stream>>>(pos, nbr);
  lin0_kernel<<<(Np * 64) / 256, 256, 0, stream>>>(pos, lin0_w, lin0_b, x0, xaltbf);

  run_block(x0, xaltbf, 64, 64,   LP[0], 0, x1, x1bf);
  run_block(x1, x1bf,   64, 128,  LP[1], 1, x2, xaltbf);
  run_block(x2, xaltbf, 128, 128, LP[2], 2, x3, x3bf);
  run_block(x3, x3bf,   128, 256, LP[3], 3, x4, nullptr);

  // head: concat(x1,x3,x4) -> 448 -> relu(512) -> 128 -> arcface
  concat_kernel<<<(unsigned)(((size_t)Np * 448) / 256), 256, 0, stream>>>(
      x1, x3, x4, combbf);
  gemm(combbf, hw1_bf, head_b1, nullptr, h512bf, 448, 512, 1);
  gemm(h512bf, hw2_bf, head_b2, emb, nullptr, 512, 128, 0);
  normw_kernel<<<1, 128, 0, stream>>>(arc_w, wn);
  arcface_kernel<<<Np / 4, 128, 0, stream>>>(emb, wn, label, (float*)d_out);
}
